// Informer_27900107555235
// MI455X (gfx1250) — matrix operand order repaired and checked by reading, not compiled, not run on hardware
//
#include <hip/hip_runtime.h>
#include <math.h>
#include <stdint.h>

#define THREADS 256

typedef __attribute__((ext_vector_type(16))) __bf16 v16bf;
typedef __attribute__((ext_vector_type(8)))  float  v8f;
typedef __attribute__((ext_vector_type(4)))  float  f4;
typedef __attribute__((ext_vector_type(8)))  unsigned short us8;

union Frag { us8 h[2]; v16bf v; };

__device__ __forceinline__ unsigned short f2bf(float x) {
  unsigned int u = __builtin_bit_cast(unsigned int, x);
  u += 0x7FFFu + ((u >> 16) & 1u);          // round-to-nearest-even
  return (unsigned short)(u >> 16);
}

// ---------------------------------------------------------------------------
// Universal batched WMMA GEMM:  C = epilogue( alpha * A@B + bias )
// Block tile 64x64, K-step 32; 8 wave32 waves, each wave owns 16(M)x32(N)
// -> 2 x v_wmma_f32_16x16x32_bf16 per K-step per wave.
// A panel is staged fp32 into LDS via GLOBAL_LOAD_ASYNC_TO_LDS_B128 (ASYNCcnt),
// converted to bf16 during fragment build (co-executes with the XDL pipe).
// B panel is staged bf16, transposed to Bs[n][k] for b128 fragment loads.
// Batch z: bh = bh0+z; b = bh/H; h = bh%H
//   A += aLocal ? z*sAb : b*sAb + h*sAh ;  B += b*sBb + h*sBh ;  C += (cb0+z)*sC
// act: 0 none, 1 exact GELU, 2 distil scale-affine(gamma,beta)+ELU
// ---------------------------------------------------------------------------
__global__ __launch_bounds__(THREADS)
void k_gemm(const float* __restrict__ A, const float* __restrict__ Bm,
            const float* __restrict__ bias, const float* __restrict__ gamma,
            const float* __restrict__ beta, float* __restrict__ C,
            int M, int N, int K, int lda, int ldb, int ldc,
            long long sAb, long long sAh, long long sBb, long long sBh, long long sC,
            int H, int bh0, int cb0, int aLocal, int transB, float alpha, int act)
{
  __shared__ __align__(16) float          Af[64][36];   // 64(m) x 32(k) fp32 (+pad)
  __shared__ __align__(16) unsigned short Bs[64][40];   // 64(n) x 32(k) bf16 transposed

  const int z  = blockIdx.z;
  const int bh = bh0 + z;
  const int b  = bh / H;
  const int hh = bh - b * H;
  A  += aLocal ? (long long)z * sAb : ((long long)b * sAb + (long long)hh * sAh);
  Bm += (long long)b * sBb + (long long)hh * sBh;
  C  += (long long)(cb0 + z) * sC;

  const int row0 = blockIdx.y * 64;
  const int col0 = blockIdx.x * 64;
  const int tid  = threadIdx.x;
  const int lane = tid & 31;
  const int w    = tid >> 5;        // 8 waves
  const int wm   = w >> 1;          // 4 along M (16 rows each)
  const int wn   = (w & 1) * 32;    // 2 along N (32 cols each)
  const int g    = lane >> 4;
  const int l15  = lane & 15;

  const bool a16 = ((((uintptr_t)A) & 15u) == 0) && ((lda & 3) == 0);
  const bool b16 = ((((uintptr_t)Bm) & 15u) == 0) && ((ldb & 3) == 0);
  const bool aFullMN = (row0 + 64 <= M);
  const bool bFullMN = (col0 + 64 <= N);

  v8f acc0 = {0.f,0.f,0.f,0.f,0.f,0.f,0.f,0.f};
  v8f acc1 = {0.f,0.f,0.f,0.f,0.f,0.f,0.f,0.f};

  for (int k0 = 0; k0 < K; k0 += 32) {
    const bool kfull = (k0 + 32 <= K);
    // ---- stage A tile (64 x 32 fp32) ---------------------------------------
    if (kfull && aFullMN) {
      const int r = tid >> 2, kk0 = (tid & 3) * 8;
      const float* ap = A + (long long)(row0 + r) * lda + (k0 + kk0);
      if (a16) {
        // async DMA: 2 x 16B, global->LDS, no VGPR round-trip (ASYNCcnt)
        unsigned lo = (unsigned)(uintptr_t)&Af[r][kk0];
        asm volatile("global_load_async_to_lds_b128 %0, %1, off"
                     :: "v"(lo), "v"(ap) : "memory");
        asm volatile("global_load_async_to_lds_b128 %0, %1, off offset:16"
                     :: "v"(lo), "v"(ap) : "memory");
      } else {
        f4 v0, v1;
#pragma unroll
        for (int j = 0; j < 4; ++j) { v0[j] = ap[j]; v1[j] = ap[4 + j]; }
        *(f4*)&Af[r][kk0]     = v0;
        *(f4*)&Af[r][kk0 + 4] = v1;
      }
    } else {
#pragma unroll
      for (int i = 0; i < 8; ++i) {
        int e = tid + THREADS * i;
        int r = e >> 5, kk = e & 31;
        int gm = row0 + r, gk = k0 + kk;
        Af[r][kk] = (gm < M && gk < K) ? A[(long long)gm * lda + gk] : 0.f;
      }
    }
    // ---- stage B tile bf16 transposed: Bs[n][k] ----------------------------
    const bool bfull = kfull && bFullMN;
    if (transB) {
      if (bfull) {
        const int n = tid >> 2, kk0 = (tid & 3) * 8;
        const float* bp = Bm + (long long)(col0 + n) * ldb + (k0 + kk0);
        us8 pk;
        if (b16) {
          f4 v0 = *(const f4*)bp;
          f4 v1 = *(const f4*)(bp + 4);
#pragma unroll
          for (int j = 0; j < 4; ++j) { pk[j] = f2bf(v0[j]); pk[j + 4] = f2bf(v1[j]); }
        } else {
#pragma unroll
          for (int j = 0; j < 8; ++j) pk[j] = f2bf(bp[j]);
        }
        *(us8*)&Bs[n][kk0] = pk;
      } else {
#pragma unroll
        for (int i = 0; i < 8; ++i) {
          int e = tid + THREADS * i;
          int kk = e >> 6, n = e & 63;
          int gn = col0 + n, gk = k0 + kk;
          float v = (gn < N && gk < K) ? Bm[(long long)gn * ldb + gk] : 0.f;
          Bs[n][kk] = f2bf(v);
        }
      }
    } else {
      if (bfull) {
        const int k = tid >> 3, n0 = (tid & 7) * 8;
        const float* bp = Bm + (long long)(k0 + k) * ldb + (col0 + n0);
        if (b16) {
          f4 v0 = *(const f4*)bp;
          f4 v1 = *(const f4*)(bp + 4);
#pragma unroll
          for (int j = 0; j < 4; ++j) { Bs[n0 + j][k] = f2bf(v0[j]); Bs[n0 + 4 + j][k] = f2bf(v1[j]); }
        } else {
#pragma unroll
          for (int j = 0; j < 8; ++j) Bs[n0 + j][k] = f2bf(bp[j]);
        }
      } else {
#pragma unroll
        for (int i = 0; i < 8; ++i) {
          int e = tid + THREADS * i;
          int kk = e >> 6, n = e & 63;
          int gn = col0 + n, gk = k0 + kk;
          float v = (gn < N && gk < K) ? Bm[(long long)gk * ldb + gn] : 0.f;
          Bs[n][kk] = f2bf(v);
        }
      }
    }
    // wait for this wave's async LDS writes, then workgroup barrier
    asm volatile("s_wait_asynccnt 0x0" ::: "memory");
    __syncthreads();

    // ---- fragments per ISA 7.12.2 + 2x WMMA --------------------------------
    Frag fa, fb0, fb1;
    {
      const int am = wm * 16 + l15;
      f4 x0 = *(const f4*)&Af[am][g * 8];
      f4 x1 = *(const f4*)&Af[am][g * 8 + 4];
      f4 x2 = *(const f4*)&Af[am][16 + g * 8];
      f4 x3 = *(const f4*)&Af[am][16 + g * 8 + 4];
#pragma unroll
      for (int j = 0; j < 4; ++j) {
        fa.v[j]      = (__bf16)x0[j];
        fa.v[4 + j]  = (__bf16)x1[j];
        fa.v[8 + j]  = (__bf16)x2[j];
        fa.v[12 + j] = (__bf16)x3[j];
      }
    }
    const int bn0 = wn + l15;
    fb0.h[0] = *(const us8*)&Bs[bn0][g * 8];
    fb0.h[1] = *(const us8*)&Bs[bn0][16 + g * 8];
    const int bn1 = wn + 16 + l15;
    fb1.h[0] = *(const us8*)&Bs[bn1][g * 8];
    fb1.h[1] = *(const us8*)&Bs[bn1][16 + g * 8];

    acc0 = __builtin_amdgcn_wmma_f32_16x16x32_bf16(false, fa.v, false, fb0.v,
                                                   (short)0, acc0, false, false);
    acc1 = __builtin_amdgcn_wmma_f32_16x16x32_bf16(false, fa.v, false, fb1.v,
                                                   (short)0, acc1, false, false);
    __syncthreads();
  }

  // ---- epilogue: element (m = r + 8*g, n = l15) within each 16x16 tile -----
#pragma unroll
  for (int j = 0; j < 2; ++j) {
    v8f acc = j ? acc1 : acc0;
#pragma unroll
    for (int r = 0; r < 8; ++r) {
      int m = row0 + wm * 16 + r + 8 * g;
      int n = col0 + wn + j * 16 + l15;
      if (m < M && n < N) {
        float v = acc[r] * alpha;
        if (bias) v += bias[n];
        if (act == 2) {
          v *= 0.99999500003749969f;               // 1/sqrt(1+1e-5)
          v = v * gamma[n] + beta[n];
          v = v > 0.f ? v : (__expf(v) - 1.f);     // ELU
        } else if (act == 1) {
          v = 0.5f * v * (1.f + erff(v * 0.70710678118654752f));  // exact GELU
        }
        C[(long long)m * ldc + n] = v;
      }
    }
  }
}

// ---------------------------------------------------------------------------
// Row softmax with optional causal mask (masked tail zeroed). One block/row.
// ---------------------------------------------------------------------------
__global__ __launch_bounds__(THREADS)
void k_softmax(float* __restrict__ S, int Lq, int Lk, int causal)
{
  long long row = blockIdx.x;
  int q = (int)(row % Lq);
  float* p = S + row * (long long)Lk;
  int lim = causal ? (q + 1) : Lk;
  int tid = threadIdx.x;
  __shared__ float red[THREADS];

  float mx = -3.4e38f;
  for (int j = tid; j < lim; j += THREADS) mx = fmaxf(mx, p[j]);
  red[tid] = mx; __syncthreads();
  for (int o = THREADS / 2; o; o >>= 1) { if (tid < o) red[tid] = fmaxf(red[tid], red[tid + o]); __syncthreads(); }
  float gm = red[0]; __syncthreads();

  float s = 0.f;
  for (int j = tid; j < lim; j += THREADS) { float e = __expf(p[j] - gm); p[j] = e; s += e; }
  red[tid] = s; __syncthreads();
  for (int o = THREADS / 2; o; o >>= 1) { if (tid < o) red[tid] += red[tid + o]; __syncthreads(); }
  float inv = 1.f / red[0];

  for (int j = tid; j < lim; j += THREADS) p[j] *= inv;
  for (int j = lim + tid; j < Lk; j += THREADS) p[j] = 0.f;
}

// residual add + layernorm (r may be null; in-place out==x allowed)
__global__ __launch_bounds__(THREADS)
void k_addln(const float* __restrict__ x, const float* __restrict__ r,
             const float* __restrict__ g, const float* __restrict__ b,
             float* __restrict__ out, int D)
{
  long long row = blockIdx.x;
  x += row * D; if (r) r += row * D; out += row * D;
  int tid = threadIdx.x;
  __shared__ float red[THREADS];

  float s = 0.f;
  for (int c = tid; c < D; c += THREADS) s += x[c] + (r ? r[c] : 0.f);
  red[tid] = s; __syncthreads();
  for (int o = THREADS / 2; o; o >>= 1) { if (tid < o) red[tid] += red[tid + o]; __syncthreads(); }
  float mean = red[0] / D; __syncthreads();

  float s2 = 0.f;
  for (int c = tid; c < D; c += THREADS) { float v = x[c] + (r ? r[c] : 0.f) - mean; s2 += v * v; }
  red[tid] = s2; __syncthreads();
  for (int o = THREADS / 2; o; o >>= 1) { if (tid < o) red[tid] += red[tid + o]; __syncthreads(); }
  float inv = rsqrtf(red[0] / D + 1e-5f); __syncthreads();

  for (int c = tid; c < D; c += THREADS) {
    float v = x[c] + (r ? r[c] : 0.f);
    out[c] = (v - mean) * inv * g[c] + b[c];
  }
}

// merge heads: ctx (B,H,Lq,dh) -> out (B,Lq,H*dh); mix = informer reshape quirk
__global__ __launch_bounds__(THREADS)
void k_merge(const float* __restrict__ ctx, float* __restrict__ out,
             int Lq, int dh, int H, int mix, long long total)
{
  long long i = (long long)blockIdx.x * THREADS + threadIdx.x;
  if (i >= total) return;
  int D = H * dh;
  long long per = (long long)Lq * D;
  long long b = i / per;
  long long rem = i - b * per;        // = l*D + c
  long long src;
  if (mix) {
    long long h0 = rem / ((long long)Lq * dh);
    long long r2 = rem - h0 * (long long)Lq * dh;
    long long l0 = r2 / dh, d0 = r2 - l0 * dh;
    src = ((b * H + h0) * Lq + l0) * dh + d0;
  } else {
    long long l = rem / D, c = rem - l * D;
    long long h0 = c / dh, d0 = c - h0 * dh;
    src = ((b * H + h0) * Lq + l) * dh + d0;
  }
  out[i] = ctx[src];
}

// circular-pad im2col: x (B,Lc,Cin) -> out (B*Lc, 3*Cin)
__global__ __launch_bounds__(THREADS)
void k_im2col(const float* __restrict__ x, float* __restrict__ out,
              int Lc, int Cin, long long total)
{
  long long i = (long long)blockIdx.x * THREADS + threadIdx.x;
  if (i >= total) return;
  int K3 = 3 * Cin;
  long long b = i / ((long long)Lc * K3);
  long long rem = i - b * (long long)Lc * K3;
  int l = (int)(rem / K3);
  int k = (int)(rem - (long long)l * K3);
  int t = k / Cin, ci = k - t * Cin;
  int ls = l + t - 1; ls = (ls + Lc) % Lc;
  out[i] = x[((long long)b * Lc + ls) * Cin + ci];
}

// sinusoidal positional add (D=512)
__global__ __launch_bounds__(THREADS)
void k_addpos(float* __restrict__ x, int Lc, int D, long long total)
{
  long long i = (long long)blockIdx.x * THREADS + threadIdx.x;
  if (i >= total) return;
  int c = (int)(i % D);
  int l = (int)((i / D) % Lc);
  int i2 = c & ~1;
  float div = __expf(-(float)i2 * (9.210340371976184f / (float)D));
  float ang = (float)l * div;
  x[i] += (c & 1) ? cosf(ang) : sinf(ang);
}

// maxpool k=3 s=2 pad=1 over length dim: (B,Lin,C) -> (B,Lin/2,C)
__global__ __launch_bounds__(THREADS)
void k_maxpool(const float* __restrict__ x, float* __restrict__ y,
               int Lin, int C, long long total)
{
  long long i = (long long)blockIdx.x * THREADS + threadIdx.x;
  if (i >= total) return;
  int Lout = Lin / 2;
  int c = (int)(i % C);
  long long bl = i / C;
  int lo = (int)(bl % Lout);
  long long b = bl / Lout;
  float m = -3.4e38f;
#pragma unroll
  for (int t = -1; t <= 1; ++t) {
    int l = 2 * lo + t;
    if (l >= 0 && l < Lin) m = fmaxf(m, x[((long long)b * Lin + l) * C + c]);
  }
  y[i] = m;
}

// Y/U gathers: yu (BL,80) = [app_embed[app] | app_embed1[user]]; y (BL,64) copy
__global__ __launch_bounds__(THREADS)
void k_yu(const int* __restrict__ app, const int* __restrict__ usr,
          const float* __restrict__ emb, const float* __restrict__ emb1,
          float* __restrict__ yu, float* __restrict__ y, long long total)
{
  long long i = (long long)blockIdx.x * THREADS + threadIdx.x;
  if (i >= total) return;
  long long row = i / 80;
  int c = (int)(i - row * 80);
  float v;
  if (c < 64) { v = emb[(long long)app[row] * 64 + c]; y[row * 64 + c] = v; }
  else        { v = emb1[(long long)usr[row] * 16 + (c - 64)]; }
  yu[i] = v;
}

// time table gather: out (BL,60)
__global__ __launch_bounds__(THREADS)
void k_time(const int* __restrict__ ts,
            const float* __restrict__ t0, const float* __restrict__ t1,
            const float* __restrict__ t2, const float* __restrict__ t3,
            const float* __restrict__ t4, float* __restrict__ out, long long total)
{
  long long i = (long long)blockIdx.x * THREADS + threadIdx.x;
  if (i >= total) return;
  long long row = i / 60;
  int c = (int)(i - row * 60);
  int t = c / 12, wi = c - t * 12;
  const float* tabs[5] = { t0, t1, t2, t3, t4 };
  int idx = ts[row * 5 + t];
  out[i] = tabs[t][(long long)idx * 12 + wi];
}

// zero-pad along length: z (B,Lin,D) -> out (B,Ltot,D)
__global__ __launch_bounds__(THREADS)
void k_padZ(const float* __restrict__ z, float* __restrict__ out,
            int Lin, int Ltot, int D, long long total)
{
  long long i = (long long)blockIdx.x * THREADS + threadIdx.x;
  if (i >= total) return;
  long long b = i / ((long long)Ltot * D);
  long long rem = i - b * (long long)Ltot * D;
  int l = (int)(rem / D);
  int c = (int)(rem - (long long)l * D);
  out[i] = (l < Lin) ? z[((long long)b * Lin + l) * D + c] : 0.f;
}

// ===========================================================================
// Host orchestration
// ===========================================================================
static inline int cdiv(int a, int b) { return (a + b - 1) / b; }
static inline unsigned gb(long long tot) { return (unsigned)((tot + THREADS - 1) / THREADS); }

static void gemm_full(hipStream_t s, const float* A, const float* B,
                      const float* bias, const float* gamma, const float* beta,
                      float* C, int M, int N, int K, int lda, int ldb, int ldc,
                      long long sAb, long long sAh, long long sBb, long long sBh,
                      long long sC, int H, int bh0, int cb0, int aLocal,
                      int batch, int transB, float alpha, int act)
{
  dim3 gr((unsigned)cdiv(N, 64), (unsigned)cdiv(M, 64), (unsigned)batch);
  k_gemm<<<gr, THREADS, 0, s>>>(A, B, bias, gamma, beta, C, M, N, K, lda, ldb, ldc,
                                sAb, sAh, sBb, sBh, sC, H, bh0, cb0, aLocal,
                                transB, alpha, act);
}

static void linear(hipStream_t s, const float* A, const float* W, const float* bias,
                   float* C, int M, int N, int K, int lda, int act,
                   const float* gamma, const float* beta)
{
  gemm_full(s, A, W, bias, gamma, beta, C, M, N, K, lda, N, N,
            0, 0, 0, 0, 0, 1, 0, 0, 0, 1, 0, 1.0f, act);
}

struct Scr { float *Q, *K, *V, *ctx, *sc, *mg, *t1, *t2, *t3, *t4, *im; };

// dense multi-head attention (prob-sparse approximated by dense w/ causal mask)
static void attention(hipStream_t s, Scr& S, const float* xq, int ldq,
                      const float* xkv, int Lq, int Lk, int dm,
                      const float* Wq, const float* bq, const float* Wk, const float* bk,
                      const float* Wv, const float* bv, const float* Wo, const float* bo,
                      int causal, int mix, float* out)
{
  const int Bn = 64, Hh = 8, dh = dm / 8;
  linear(s, xq,  Wq, bq, S.Q, Bn * Lq, dm, dm, ldq, 0, nullptr, nullptr);
  linear(s, xkv, Wk, bk, S.K, Bn * Lk, dm, dm, dm,  0, nullptr, nullptr);
  linear(s, xkv, Wv, bv, S.V, Bn * Lk, dm, dm, dm,  0, nullptr, nullptr);
  const int BH = Bn * Hh, CH = 64;
  float alpha = 1.0f / sqrtf((float)dh);
  for (int bh0 = 0; bh0 < BH; bh0 += CH) {
    // scores(chunk) = alpha * Q @ K^T
    gemm_full(s, S.Q, S.K, nullptr, nullptr, nullptr, S.sc,
              Lq, Lk, dh, dm, dm, Lk,
              (long long)Lq * dm, dh, (long long)Lk * dm, dh, (long long)Lq * Lk,
              Hh, bh0, 0, 0, CH, 1, alpha, 0);
    k_softmax<<<(unsigned)(CH * Lq), THREADS, 0, s>>>(S.sc, Lq, Lk, causal);
    // ctx(bh) = A @ V
    gemm_full(s, S.sc, S.V, nullptr, nullptr, nullptr, S.ctx,
              Lq, dh, Lk, Lk, dm, dh,
              (long long)Lq * Lk, 0, (long long)Lk * dm, dh, (long long)Lq * dh,
              Hh, bh0, bh0, 1, CH, 0, 1.0f, 0);
  }
  long long tot = (long long)Bn * Lq * dm;
  k_merge<<<gb(tot), THREADS, 0, s>>>(S.ctx, S.mg, Lq, dh, Hh, mix, tot);
  linear(s, S.mg, Wo, bo, out, Bn * Lq, dm, dm, dm, 0, nullptr, nullptr);
}

// decoder layer: leaves at base: c1b,c1w,c2b,c2w, cross(Wk,Wo,Wq,Wv,bk,bo,bq,bv),
//                n1(g,b), n2(g,b), n3(g,b), self(Wk,Wo,Wq,Wv,bk,bo,bq,bv)
static void dec_layer(hipStream_t s, Scr& S, const float* const* pp, int bs,
                      float* x, const float* cross, int Lq, int Lk, int dm)
{
  const int rows = 64 * Lq;
  attention(s, S, x, dm, x, Lq, Lq, dm,
            pp[bs+20], pp[bs+24], pp[bs+18], pp[bs+22],
            pp[bs+21], pp[bs+25], pp[bs+19], pp[bs+23], 1, 1, S.t1);
  k_addln<<<(unsigned)rows, THREADS, 0, s>>>(x, S.t1, pp[bs+12], pp[bs+13], S.t2, dm);
  attention(s, S, S.t2, dm, cross, Lq, Lk, dm,
            pp[bs+6], pp[bs+10], pp[bs+4], pp[bs+8],
            pp[bs+7], pp[bs+11], pp[bs+5], pp[bs+9], 0, 0, S.t1);
  k_addln<<<(unsigned)rows, THREADS, 0, s>>>(S.t2, S.t1, pp[bs+14], pp[bs+15], S.t3, dm);
  linear(s, S.t3, pp[bs+1], pp[bs+0], S.t4, rows, 512, dm, dm, 1, nullptr, nullptr);
  linear(s, S.t4, pp[bs+3], pp[bs+2], S.t1, rows, dm, 512, 512, 0, nullptr, nullptr);
  k_addln<<<(unsigned)rows, THREADS, 0, s>>>(S.t3, S.t1, pp[bs+16], pp[bs+17], x, dm);
}

// encoder layer: leaves at base: attn(Wk,Wo,Wq,Wv,bk,bo,bq,bv), c1b,c1w,c2b,c2w,
//                n1(g,b), n2(g,b)
static void enc_layer(hipStream_t s, Scr& S, const float* const* pp, int bs,
                      float* x, int Lq)
{
  const int dm = 512, rows = 64 * Lq;
  attention(s, S, x, dm, x, Lq, Lq, dm,
            pp[bs+2], pp[bs+6], pp[bs+0], pp[bs+4],
            pp[bs+3], pp[bs+7], pp[bs+1], pp[bs+5], 0, 0, S.t1);
  k_addln<<<(unsigned)rows, THREADS, 0, s>>>(x, S.t1, pp[bs+12], pp[bs+13], S.t2, dm);
  linear(s, S.t2, pp[bs+9], pp[bs+8], S.t3, rows, 512, dm, dm, 1, nullptr, nullptr);
  linear(s, S.t3, pp[bs+11], pp[bs+10], S.t1, rows, dm, 512, 512, 0, nullptr, nullptr);
  k_addln<<<(unsigned)rows, THREADS, 0, s>>>(S.t2, S.t1, pp[bs+14], pp[bs+15], x, dm);
}

static void dec1_stack(hipStream_t s, Scr& S, const float* const* pp,
                       float* x, const float* cross)
{
  dec_layer(s, S, pp, 7,  x, cross, 336, 336, 64);
  dec_layer(s, S, pp, 33, x, cross, 336, 336, 64);
  k_addln<<<(unsigned)(64 * 336), THREADS, 0, s>>>(x, nullptr, pp[59], pp[60], x, 64);
}

extern "C" void kernel_launch(void* const* d_in, const int* in_sizes, int n_in,
                              void* d_out, int out_size, void* d_ws, size_t ws_size,
                              hipStream_t stream)
{
  (void)in_sizes; (void)n_in; (void)out_size; (void)ws_size;
  // d_in: 0 app_seq_pred(i32), 1 time_seq(i32), 2 location_vectors(f32),
  //       3 user(i32), 4.. params leaves in jax pytree order (dicts key-sorted)
  const int*   app = (const int*)d_in[0];
  const int*   tsq = (const int*)d_in[1];
  const float* loc = (const float*)d_in[2];
  const int*   usr = (const int*)d_in[3];
  const float* pp[187];
  for (int i = 4; i < 187; ++i) pp[i] = (const float*)d_in[i];
  hipStream_t s = stream;

  char* base = (char*)d_ws; size_t off = 0;
  auto alloc = [&](long long nf) -> float* {
    float* p = (float*)(base + off);
    off += (size_t)((nf * 4 + 255) / 256) * 256;
    return p;
  };

  const int Bn = 64, Ll = 336, Lp = 504;
  const long long BL = (long long)Bn * Ll, BLp = (long long)Bn * Lp;

  Scr S;
  S.Q   = alloc(BLp * 512); S.K  = alloc(BLp * 512); S.V = alloc(BLp * 512);
  S.ctx = alloc(BLp * 512); S.mg = alloc(BLp * 512);
  S.sc  = alloc((long long)64 * Lp * Lp);
  S.t1  = alloc(BLp * 512); S.t2 = alloc(BLp * 512);
  S.t3  = alloc(BLp * 512); S.t4 = alloc(BLp * 512);
  S.im  = alloc(BL * 1536);
  float* Tm  = alloc(BL * 64);
  float* Lm  = alloc(BL * 64);
  float* Zb  = alloc(BL * 64);
  float* Z2  = alloc(BL * 64);
  float* Zp  = alloc(BLp * 64);
  float* yu  = alloc(BL * 80);
  float* h   = alloc(BLp * 512);
  float* enc = alloc((long long)Bn * 84 * 512);
  float* db  = alloc(BLp * 512);

  // --- input embeddings -----------------------------------------------------
  { long long tot = BL * 60;
    k_time<<<gb(tot), THREADS, 0, s>>>(tsq, pp[182], pp[183], pp[184], pp[185],
                                       pp[186], S.im, tot);
    linear(s, S.im, pp[65], pp[64], Tm, (int)BL, 64, 60, 60, 0, nullptr, nullptr); }
  linear(s, loc, pp[61], pp[6], Lm, (int)BL, 64, 17, 17, 0, nullptr, nullptr);
  { long long tot = BL * 80;
    k_yu<<<gb(tot), THREADS, 0, s>>>(app, usr, pp[4], pp[5], yu, Z2, tot); }
  linear(s, yu, pp[63], pp[62], Zb, (int)BL, 64, 80, 80, 0, nullptr, nullptr);

  // --- dec1 stacks (dm=64) --------------------------------------------------
  dec1_stack(s, S, pp, Zb, Lm);   // Z1 = dec1(X, L)
  dec1_stack(s, S, pp, Zb, Tm);   // Z1 = dec1(Z1, T)
  dec1_stack(s, S, pp, Z2, Lm);   // Z2 = dec1(Y, L)
  { long long tot = BLp * 64;
    k_padZ<<<gb(tot), THREADS, 0, s>>>(Z2, Zp, Ll, Lp, 64, tot); }

  // --- encoder --------------------------------------------------------------
  { long long tot = BL * 192;
    k_im2col<<<gb(tot), THREADS, 0, s>>>(Zb, S.im, Ll, 64, tot);
    linear(s, S.im, pp[179], nullptr, h, (int)BL, 512, 192, 192, 0, nullptr, nullptr);
    long long tp = BL * 512;
    k_addpos<<<gb(tp), THREADS, 0, s>>>(h, Ll, 512, tp); }

  int Lc = Ll;
  const int encBase[3]  = {129, 145, 161};
  const int convBase[2] = {121, 125};
  for (int i = 0; i < 3; ++i) {
    enc_layer(s, S, pp, encBase[i], h, Lc);
    if (i < 2) {
      long long rows = (long long)Bn * Lc;
      long long tot = rows * 1536;
      k_im2col<<<gb(tot), THREADS, 0, s>>>(h, S.im, Lc, 512, tot);
      linear(s, S.im, pp[convBase[i] + 3], pp[convBase[i] + 0], S.t1,
             (int)rows, 512, 1536, 1536, 2, pp[convBase[i] + 2], pp[convBase[i] + 1]);
      int Lo = Lc / 2;
      long long tp = (long long)Bn * Lo * 512;
      k_maxpool<<<gb(tp), THREADS, 0, s>>>(S.t1, h, Lc, 512, tp);
      Lc = Lo;
    }
  }
  k_addln<<<(unsigned)(Bn * Lc), THREADS, 0, s>>>(h, nullptr, pp[177], pp[178], enc, 512);

  // --- decoder (dm=512, L=504, cross=84) ------------------------------------
  { long long tot = BLp * 192;
    k_im2col<<<gb(tot), THREADS, 0, s>>>(Zp, S.im, Lp, 64, tot);
    linear(s, S.im, pp[120], nullptr, db, (int)BLp, 512, 192, 192, 0, nullptr, nullptr);
    long long tp = BLp * 512;
    k_addpos<<<gb(tp), THREADS, 0, s>>>(db, Lp, 512, tp); }
  dec_layer(s, S, pp, 66, db, enc, Lp, 84, 512);
  dec_layer(s, S, pp, 92, db, enc, Lp, 84, 512);
  k_addln<<<(unsigned)BLp, THREADS, 0, s>>>(db, nullptr, pp[118], pp[119], db, 512);

  // --- final projection: d[:, -168:] @ proj_w + proj_b -> d_out -------------
  gemm_full(s, db + 336 * 512, pp[181], pp[180], nullptr, nullptr, (float*)d_out,
            168, 2000, 512, 512, 2000, 2000,
            (long long)Lp * 512, 0, 0, 0, (long long)168 * 2000,
            1, 0, 0, 0, Bn, 0, 1.0f, 0);
}
